// HopeNetModify_90615220011219
// MI455X (gfx1250) — matrix operand order repaired and checked by reading, not compiled, not run on hardware
//
#include <hip/hip_runtime.h>
#include <cmath>

// ---------------------------------------------------------------------------
// Types for CDNA5 WMMA (wave32, 16x16x32 bf16 -> f32)
// ---------------------------------------------------------------------------
typedef __attribute__((ext_vector_type(16))) __bf16 v16bf;
typedef __attribute__((ext_vector_type(8)))  __bf16 v8bf;
typedef __attribute__((ext_vector_type(8)))  float  v8f;

__device__ __forceinline__ unsigned short f2bf(float x) {
  unsigned int u = __float_as_uint(x);
  u += 0x7FFFu + ((u >> 16) & 1u);           // round-to-nearest-even
  return (unsigned short)(u >> 16);
}

// ---------------------------------------------------------------------------
// bf16 WMMA GEMM:  C[M,N] = A[M,K] * B[K,N] (+bias)(+GELU)(+residual)
//   A  : bf16 [M,K] row-major,  K % 32 == 0, rows 16B-aligned
//   Bt : bf16 [N,K] row-major (pre-transposed weights), same K constraints
// Block tile 128x128, 8 waves: wave grid 4(M)x2(N), wave tile 32x64
// (2x4 fragments of 16x16 -> 8 WMMAs / K-step / wave).
// Staging is pure b128 traffic; M/N edges handled by index clamping
// (out-of-range rows/cols are computed but never stored).
// LDS rows padded to 40 elems (80B) -> conflict-free 16-lane fragment reads.
// ---------------------------------------------------------------------------
#define BM 128
#define BN 128
#define BK 32
#define LDP 40

template<int ACT, int OBF, int BIAS, int RES>
__global__ __launch_bounds__(256) void gemm_bf16_kernel(
    const unsigned short* __restrict__ A,
    const unsigned short* __restrict__ Bt,
    const float* __restrict__ bias,
    const float* Res,           // residual source (may alias Cv)
    void* Cv,
    int M, int N, int K)
{
  __shared__ __align__(16) unsigned short As[2][BM * LDP];
  __shared__ __align__(16) unsigned short Bs[2][BN * LDP];
  const int tid  = threadIdx.x;
  const int lane = tid & 31;
  const int w    = tid >> 5;
  const int wm   = w >> 1;     // 0..3  (32 rows each)
  const int wn   = w & 1;      // 0..1  (64 cols each)
  const int bm0  = blockIdx.y * BM;
  const int bn0  = blockIdx.x * BN;

  // staging coordinates (branch-free, clamped)
  const int arow = tid >> 1;                 // 0..127
  const int akh  = (tid & 1) * 16;           // 0 / 16
  const int agm  = min(bm0 + arow, M - 1);
  const unsigned short* aSrc = A + (size_t)agm * K + akh;
  const int brow = tid >> 1;                 // 0..127
  const int bkh  = (tid & 1) * 16;           // 0 / 16
  const int bgn  = min(bn0 + brow, N - 1);
  const unsigned short* bSrc = Bt + (size_t)bgn * K + bkh;

  v8f acc[2][4] = {};
  const int ksteps = K / BK;

  auto load_tiles = [&](int buf, int k0) {
    const uint4* as = (const uint4*)(aSrc + k0);
    uint4 a0 = as[0];
    uint4 a1 = as[1];
    const uint4* bs = (const uint4*)(bSrc + k0);
    uint4 b0 = bs[0];
    uint4 b1 = bs[1];
    if (k0 + BK < K) {
      __builtin_prefetch(aSrc + k0 + BK, 0, 3);
      __builtin_prefetch(bSrc + k0 + BK, 0, 3);
    }
    *(uint4*)&As[buf][arow * LDP + akh]     = a0;
    *(uint4*)&As[buf][arow * LDP + akh + 8] = a1;
    *(uint4*)&Bs[buf][brow * LDP + bkh]     = b0;
    *(uint4*)&Bs[buf][brow * LDP + bkh + 8] = b1;
  };

  load_tiles(0, 0);
  __syncthreads();

  const int l15 = lane & 15;
  const int khA = (lane >> 4) * 8;    // ISA A-frag: K half offset 0/8
  const int khB = (lane >> 4) * 8;    // ISA B-frag: K half offset 0/8

  for (int ks = 0; ks < ksteps; ++ks) {
    const int buf = ks & 1;
    if (ks + 1 < ksteps) load_tiles(buf ^ 1, (ks + 1) * BK);
    const __bf16* Ap = (const __bf16*)As[buf];
    const __bf16* Bp = (const __bf16*)Bs[buf];
    v16bf afr[2], bfr[4];
#pragma unroll
    for (int mf = 0; mf < 2; ++mf) {
      const int m = wm * 32 + mf * 16 + l15;
      ((v8bf*)&afr[mf])[0] = *(const v8bf*)(Ap + m * LDP + khA);       // K: kh..kh+7
      ((v8bf*)&afr[mf])[1] = *(const v8bf*)(Ap + m * LDP + 16 + khA);  // K: 16+kh..
    }
#pragma unroll
    for (int nf = 0; nf < 4; ++nf) {
      const int n = wn * 64 + nf * 16 + l15;
      ((v8bf*)&bfr[nf])[0] = *(const v8bf*)(Bp + n * LDP + khB);       // K: kh..kh+7
      ((v8bf*)&bfr[nf])[1] = *(const v8bf*)(Bp + n * LDP + 16 + khB);  // K: 16+kh..
    }
#pragma unroll
    for (int mf = 0; mf < 2; ++mf)
#pragma unroll
      for (int nf = 0; nf < 4; ++nf)
        acc[mf][nf] = __builtin_amdgcn_wmma_f32_16x16x32_bf16(
            false, afr[mf], false, bfr[nf], (short)0, acc[mf][nf], false, false);
    __syncthreads();
  }

  // Epilogue: C/D layout: VGPR r -> M = r + (lane>=16)*8, N = lane&15
  const int rh = (lane >> 4) * 8;
#pragma unroll
  for (int mf = 0; mf < 2; ++mf) {
#pragma unroll
    for (int nf = 0; nf < 4; ++nf) {
      const int col = bn0 + wn * 64 + nf * 16 + l15;
      if (col >= N) continue;
      const float bv = BIAS ? bias[col] : 0.0f;
#pragma unroll
      for (int r = 0; r < 8; ++r) {
        const int row = bm0 + wm * 32 + mf * 16 + rh + r;
        if (row >= M) continue;
        float v = acc[mf][nf][r] + bv;
        if (ACT == 1) v = 0.5f * v * (1.0f + erff(v * 0.70710678118f)); // exact GELU
        if (RES)     v += Res[(size_t)row * N + col];
        if (OBF) ((unsigned short*)Cv)[(size_t)row * N + col] = f2bf(v);
        else     ((float*)Cv)[(size_t)row * N + col] = v;
      }
    }
  }
}

// ---------------------------------------------------------------------------
// f32 [K,N] -> bf16 transposed [N,Kp] (zero-padded K..Kp), LDS-tiled
// ---------------------------------------------------------------------------
__global__ __launch_bounds__(256) void transpose_conv_kernel(
    const float* __restrict__ src, unsigned short* __restrict__ dst,
    int K, int N, int Kp)
{
  __shared__ float tile[32][33];
  const int k0 = blockIdx.x * 32, n0 = blockIdx.y * 32;
  const int tx = threadIdx.x & 31, ty = threadIdx.x >> 5;   // ty 0..7
  for (int i = ty; i < 32; i += 8) {
    const int k = k0 + i, n = n0 + tx;
    tile[i][tx] = (k < K && n < N) ? src[(size_t)k * N + n] : 0.f;
  }
  __syncthreads();
  for (int i = ty; i < 32; i += 8) {
    const int n = n0 + i, k = k0 + tx;
    if (n < N && k < Kp) dst[(size_t)n * Kp + k] = f2bf(tile[tx][i]);
  }
}

// ---------------------------------------------------------------------------
// Elementwise helpers (pairwise-packed bf16 conversion; n must be even)
// ---------------------------------------------------------------------------
__global__ void f32_to_bf16_kernel(const float2* __restrict__ s, unsigned int* __restrict__ d, size_t n2) {
  size_t i = (size_t)blockIdx.x * blockDim.x + threadIdx.x;
  const size_t st = (size_t)gridDim.x * blockDim.x;
  for (; i < n2; i += st) {
    float2 v = s[i];
    d[i] = (unsigned int)f2bf(v.x) | ((unsigned int)f2bf(v.y) << 16);
  }
}

// ---------------------------------------------------------------------------
// LayerNorm over D=768 (one block per row), eps=1e-12, optional bf16 output
// ---------------------------------------------------------------------------
template<int OBF>
__global__ __launch_bounds__(256) void ln_kernel(const float* __restrict__ X,
    const float* __restrict__ g, const float* __restrict__ b, void* __restrict__ Y, int D)
{
  const int row = blockIdx.x;
  const float* x = X + (size_t)row * D;
  __shared__ float red[2][8];
  float s = 0.f, s2 = 0.f;
  for (int i = threadIdx.x; i < D; i += blockDim.x) { float v = x[i]; s += v; s2 += v * v; }
#pragma unroll
  for (int o = 16; o > 0; o >>= 1) { s += __shfl_down(s, o, 32); s2 += __shfl_down(s2, o, 32); }
  const int wv = threadIdx.x >> 5, ln = threadIdx.x & 31;
  if (ln == 0) { red[0][wv] = s; red[1][wv] = s2; }
  __syncthreads();
  if (threadIdx.x == 0) {
    float t = 0.f, t2 = 0.f;
    for (int i = 0; i < (int)(blockDim.x >> 5); ++i) { t += red[0][i]; t2 += red[1][i]; }
    red[0][0] = t; red[1][0] = t2;
  }
  __syncthreads();
  const float mean = red[0][0] / D;
  const float var  = red[1][0] / D - mean * mean;
  const float inv  = rsqrtf(var + 1e-12f);
  for (int i = threadIdx.x; i < D; i += blockDim.x) {
    float v = (x[i] - mean) * inv * g[i] + b[i];
    if (OBF) ((unsigned short*)Y)[(size_t)row * D + i] = f2bf(v);
    else     ((float*)Y)[(size_t)row * D + i] = v;
  }
}

// ---------------------------------------------------------------------------
// ViT patchify: x[B,3,224,224] -> bf16 patches [B*196, 768] (c,py,px order)
// ---------------------------------------------------------------------------
__global__ void patchify_kernel(const float* __restrict__ x, unsigned short* __restrict__ out) {
  const int bp = blockIdx.x;              // b*196 + ph*14 + pw
  const int b = bp / 196, p = bp % 196;
  const int ph = p / 14, pw = p % 14;
  for (int col = threadIdx.x; col < 768; col += blockDim.x) {
    const int c = col >> 8, r = col & 255;
    const int py = r >> 4, px = r & 15;
    float v = x[(((size_t)b * 3 + c) * 224 + (ph * 16 + py)) * 224 + (pw * 16 + px)];
    out[(size_t)bp * 768 + col] = f2bf(v);
  }
}

// h[b,t,:] = (t==0 ? cls : patch_embed[b,t-1]) + pos[t]   (bias added in GEMM)
__global__ void assemble_h_kernel(const float* __restrict__ pe, const float* __restrict__ cls,
                                  const float* __restrict__ pos, float* __restrict__ h) {
  const int row = blockIdx.x;             // b*197 + t
  const int b = row / 197, t = row % 197;
  float* hp = h + (size_t)row * 768;
  const float* pp = pos + (size_t)t * 768;
  if (t == 0) {
    for (int i = threadIdx.x; i < 768; i += blockDim.x) hp[i] = cls[i] + pp[i];
  } else {
    const float* sp = pe + ((size_t)b * 196 + (t - 1)) * 768;
    for (int i = threadIdx.x; i < 768; i += blockDim.x) hp[i] = sp[i] + pp[i];
  }
}

// ---------------------------------------------------------------------------
// Attention: one wave per (b,h,q) row; scores in LDS; softmax; AV.
// Q/K/V/O layout [B,S,H*HD] with head-major columns (h*64+d).
// ---------------------------------------------------------------------------
__global__ __launch_bounds__(256) void attn_kernel(
    const float* __restrict__ Q, const float* __restrict__ Km, const float* __restrict__ V,
    float* __restrict__ O, int S, int H)
{
  const int bh = blockIdx.x;
  const int b = bh / H, h = bh % H;
  const int wv = threadIdx.x >> 5, lane = threadIdx.x & 31;
  const int q = blockIdx.y * 8 + wv;
  __shared__ float sc[8][224];
  __shared__ float sq[8][64];
  if (q >= S) return;
  const size_t stT = (size_t)H * 64;
  const float* qp = Q + ((size_t)b * S + q) * stT + h * 64;
  sq[wv][lane] = qp[lane];
  sq[wv][lane + 32] = qp[lane + 32];
  float mx = -1e30f;
  for (int k = lane; k < S; k += 32) {
    const float* kp = Km + ((size_t)b * S + k) * stT + h * 64;
    float d = 0.f;
#pragma unroll 8
    for (int i = 0; i < 64; ++i) d += sq[wv][i] * kp[i];
    d *= 0.125f;                      // 1/sqrt(64)
    sc[wv][k] = d;
    mx = fmaxf(mx, d);
  }
#pragma unroll
  for (int o = 16; o > 0; o >>= 1) mx = fmaxf(mx, __shfl_xor(mx, o, 32));
  float sum = 0.f;
  for (int k = lane; k < S; k += 32) {
    float e = expf(sc[wv][k] - mx);
    sc[wv][k] = e;
    sum += e;
  }
#pragma unroll
  for (int o = 16; o > 0; o >>= 1) sum += __shfl_xor(sum, o, 32);
  const float inv = 1.f / sum;
  float o0 = 0.f, o1 = 0.f;
  for (int k = 0; k < S; ++k) {
    const float* vp = V + ((size_t)b * S + k) * stT + h * 64;
    const float a = sc[wv][k];
    o0 += a * vp[lane];
    o1 += a * vp[lane + 32];
  }
  float* op = O + ((size_t)b * S + q) * stT + h * 64;
  op[lane] = o0 * inv;
  op[lane + 32] = o1 * inv;
}

// mean over 197 tokens
__global__ void mean_kernel(const float* __restrict__ X, float* __restrict__ F) {
  const int b = blockIdx.x;
  for (int d = threadIdx.x; d < 768; d += blockDim.x) {
    float s = 0.f;
    for (int t = 0; t < 197; ++t) s += X[((size_t)b * 197 + t) * 768 + d];
    F[b * 768 + d] = s * (1.0f / 197.0f);
  }
}

// head fc: pts_init[b, j] = feats[b] . W[:,j] + b[j]   (j < 58)
__global__ void head_kernel(const float* __restrict__ F, const float* __restrict__ W,
                            const float* __restrict__ bb, float* __restrict__ out) {
  const int idx = blockIdx.x * blockDim.x + threadIdx.x;
  if (idx >= 32 * 58) return;
  const int b = idx / 58, j = idx % 58;
  float s = bb[j];
  for (int k = 0; k < 768; ++k) s += F[b * 768 + k] * W[k * 58 + j];
  out[idx] = s;
}

// node_in bf16: [B*29, 800] (2 coords + 768 feats + 30 zero pad)
__global__ void nodein_kernel(const float* __restrict__ ptsInit, const float* __restrict__ F,
                              unsigned short* __restrict__ NIbf) {
  size_t i = (size_t)blockIdx.x * blockDim.x + threadIdx.x;
  const size_t total = 928ull * 800ull;
  const size_t st = (size_t)gridDim.x * blockDim.x;
  for (; i < total; i += st) {
    const int row = (int)(i / 800), c = (int)(i % 800);
    const int b = row / 29, n = row % 29;
    float v = 0.f;
    if (c < 2) v = ptsInit[(b * 29 + n) * 2 + c];
    else if (c < 770) v = F[b * 768 + (c - 2)];
    NIbf[i] = f2bf(v);
  }
}

// ---------------------------------------------------------------------------
// GAT aggregation (per sample): additive attention + segment softmax over dst
// ---------------------------------------------------------------------------
__global__ __launch_bounds__(256) void gat_agg_kernel(
    const float* __restrict__ Hf, const float* __restrict__ att_src, const float* __restrict__ att_dst,
    const float* __restrict__ bias, const int* __restrict__ esrc, const int* __restrict__ edst,
    int E, int Hh, int C, float* __restrict__ Out)
{
  const int N = 29;
  const int b = blockIdx.x, tid = threadIdx.x, nt = blockDim.x;
  const int F = Hh * C;
  const float* Hb = Hf + (size_t)b * N * F;
  float* Ob = Out + (size_t)b * N * F;
  __shared__ float s_as[29 * 4], s_ad[29 * 4], s_e[93 * 4], s_m[29 * 4], s_s[29 * 4];
  __shared__ int s_src[93], s_dst[93];
  for (int e = tid; e < E; e += nt) { s_src[e] = esrc[e]; s_dst[e] = edst[e]; }
  __syncthreads();
  for (int t = tid; t < N * Hh; t += nt) {
    const int n = t / Hh, h = t % Hh;
    float a1 = 0.f, a2 = 0.f;
    const float* hp = Hb + (size_t)n * F + h * C;
    for (int c = 0; c < C; ++c) { float v = hp[c]; a1 += v * att_src[h * C + c]; a2 += v * att_dst[h * C + c]; }
    s_as[t] = a1; s_ad[t] = a2;
  }
  __syncthreads();
  for (int t = tid; t < E * Hh; t += nt) {
    const int e = t / Hh, h = t % Hh;
    float a = s_as[s_src[e] * Hh + h] + s_ad[s_dst[e] * Hh + h];
    s_e[t] = (a > 0.f) ? a : 0.2f * a;                 // leaky_relu 0.2
  }
  __syncthreads();
  for (int t = tid; t < N * Hh; t += nt) {
    const int n = t / Hh, h = t % Hh;
    float m = -1e30f;
    for (int e = 0; e < E; ++e) if (s_dst[e] == n) m = fmaxf(m, s_e[e * Hh + h]);
    s_m[t] = m;
  }
  __syncthreads();
  for (int t = tid; t < E * Hh; t += nt) {
    const int e = t / Hh, h = t % Hh;
    s_e[t] = expf(s_e[t] - s_m[s_dst[e] * Hh + h]);
  }
  __syncthreads();
  for (int t = tid; t < N * Hh; t += nt) {
    const int n = t / Hh, h = t % Hh;
    float s = 0.f;
    for (int e = 0; e < E; ++e) if (s_dst[e] == n) s += s_e[e * Hh + h];
    s_s[t] = s;
  }
  __syncthreads();
  for (int idx = tid; idx < N * F; idx += nt) {
    const int n = idx / F, rem = idx % F, h = rem / C;
    float a = 0.f;
    for (int e = 0; e < E; ++e)
      if (s_dst[e] == n) a += s_e[e * Hh + h] * Hb[(size_t)s_src[e] * F + rem];
    Ob[idx] = a / s_s[n * Hh + h] + bias[rem];
  }
}

// ---------------------------------------------------------------------------
// Graph U-Net (tiny) — one block per sample, everything in LDS
// ---------------------------------------------------------------------------
struct UnetParams {
  const float* A[5];
  const float* gW[10]; const float* gB[10];
  const float* pW[5];  const float* pB[5];
  const float* uW[5];  const float* uB[5];
  const float* fc1W; const float* fc1b; const float* fc2W; const float* fc2b;
};

__device__ void u_pool(const float* __restrict__ P, const float* __restrict__ bb,
                       const float* __restrict__ Xin, float* __restrict__ Xout,
                       int m, int n, int f, int tid, int nt) {
  for (int idx = tid; idx < m * f; idx += nt) {
    const int i = idx / f, c = idx % f;
    float s = 0.f;
    for (int j = 0; j < n; ++j) s += P[i * n + j] * Xin[j * f + c];
    Xout[idx] = s + bb[i];
  }
  __syncthreads();
}
__device__ void u_concat(const float* Xa, int fa, const float* Xb, int fb,
                         float* Xo, int n, int tid, int nt) {
  const int f = fa + fb;
  for (int idx = tid; idx < n * f; idx += nt) {
    const int i = idx / f, c = idx % f;
    Xo[idx] = (c < fa) ? Xa[i * fa + c] : Xb[i * fb + (c - fa)];
  }
  __syncthreads();
}
__device__ void u_gconv(const float* __restrict__ A, const float* __restrict__ W,
                        const float* __restrict__ bb, const float* __restrict__ Xin,
                        float* __restrict__ Z, float* __restrict__ Dv, float* __restrict__ Xout,
                        int n, int fin, int fout, int relu, int tid, int nt) {
  for (int i = tid; i < n; i += nt) {
    float s = 0.f;
    for (int j = 0; j < n; ++j) s += A[i * n + j];
    Dv[i] = rsqrtf(s + 1e-5f);
  }
  __syncthreads();
  for (int idx = tid; idx < n * fin; idx += nt) {
    const int i = idx / fin, c = idx % fin;
    float s = 0.f;
    for (int j = 0; j < n; ++j) s += Dv[i] * A[i * n + j] * Dv[j] * Xin[j * fin + c];
    Z[idx] = s;
  }
  __syncthreads();
  for (int idx = tid; idx < n * fout; idx += nt) {
    const int i = idx / fout, o = idx % fout;
    float s = bb[o];
    for (int c = 0; c < fin; ++c) s += Z[i * fin + c] * W[c * fout + o];
    Xout[idx] = relu ? fmaxf(s, 0.f) : s;
  }
  __syncthreads();
}

__global__ __launch_bounds__(128) void unet_kernel(const float* __restrict__ X2d, UnetParams up,
                                                   float* __restrict__ out3d)
{
  const int b = blockIdx.x, tid = threadIdx.x, nt = blockDim.x;
  __shared__ float sX0[29 * 4], sX1[15 * 8], sX2[7 * 16], sX3[4 * 32], sX4[2 * 64];
  __shared__ float sZ[320], sC[320], sP[320], sD[32];
  __shared__ float sX5[64], sG[32], sG2[64], sYa[96], sYb[96];

  for (int i = tid; i < 29 * 2; i += nt) sC[i] = X2d[b * 58 + i];
  __syncthreads();
  u_gconv(up.A[0], up.gW[0], up.gB[0], sC, sZ, sD, sX0, 29, 2, 4, 1, tid, nt);
  u_pool(up.pW[0], up.pB[0], sX0, sP, 15, 29, 4, tid, nt);
  u_gconv(up.A[1], up.gW[1], up.gB[1], sP, sZ, sD, sX1, 15, 4, 8, 1, tid, nt);
  u_pool(up.pW[1], up.pB[1], sX1, sP, 7, 15, 8, tid, nt);
  u_gconv(up.A[2], up.gW[2], up.gB[2], sP, sZ, sD, sX2, 7, 8, 16, 1, tid, nt);
  u_pool(up.pW[2], up.pB[2], sX2, sP, 4, 7, 16, tid, nt);
  u_gconv(up.A[3], up.gW[3], up.gB[3], sP, sZ, sD, sX3, 4, 16, 32, 1, tid, nt);
  u_pool(up.pW[3], up.pB[3], sX3, sP, 2, 4, 32, tid, nt);
  u_gconv(up.A[4], up.gW[4], up.gB[4], sP, sZ, sD, sX4, 2, 32, 64, 1, tid, nt);
  u_pool(up.pW[4], up.pB[4], sX4, sX5, 1, 2, 64, tid, nt);
  for (int o = tid; o < 20; o += nt) {
    float s = up.fc1b[o];
    for (int c = 0; c < 64; ++c) s += sX5[c] * up.fc1W[c * 20 + o];
    sG[o] = fmaxf(s, 0.f);
  }
  __syncthreads();
  for (int o = tid; o < 64; o += nt) {
    float s = up.fc2b[o];
    for (int c = 0; c < 20; ++c) s += sG[c] * up.fc2W[c * 64 + o];
    sG2[o] = fmaxf(s, 0.f);
  }
  __syncthreads();
  u_pool(up.uW[0], up.uB[0], sG2, sP, 2, 1, 64, tid, nt);
  u_concat(sX4, 64, sP, 64, sC, 2, tid, nt);
  u_gconv(up.A[4], up.gW[5], up.gB[5], sC, sZ, sD, sYa, 2, 128, 32, 1, tid, nt);
  u_pool(up.uW[1], up.uB[1], sYa, sP, 4, 2, 32, tid, nt);
  u_concat(sX3, 32, sP, 32, sC, 4, tid, nt);
  u_gconv(up.A[3], up.gW[6], up.gB[6], sC, sZ, sD, sYb, 4, 64, 16, 1, tid, nt);
  u_pool(up.uW[2], up.uB[2], sYb, sP, 7, 4, 16, tid, nt);
  u_concat(sX2, 16, sP, 16, sC, 7, tid, nt);
  u_gconv(up.A[2], up.gW[7], up.gB[7], sC, sZ, sD, sYa, 7, 32, 8, 1, tid, nt);
  u_pool(up.uW[3], up.uB[3], sYa, sP, 15, 7, 8, tid, nt);
  u_concat(sX1, 8, sP, 8, sC, 15, tid, nt);
  u_gconv(up.A[1], up.gW[8], up.gB[8], sC, sZ, sD, sYb, 15, 16, 4, 1, tid, nt);
  u_pool(up.uW[4], up.uB[4], sYb, sP, 29, 15, 4, tid, nt);
  u_concat(sX0, 4, sP, 4, sC, 29, tid, nt);
  u_gconv(up.A[0], up.gW[9], up.gB[9], sC, sZ, sD, sYa, 29, 8, 3, 0, tid, nt);
  for (int i = tid; i < 29 * 3; i += nt) out3d[b * 87 + i] = sYa[i];
}

// ---------------------------------------------------------------------------
// Host orchestration
// ---------------------------------------------------------------------------
extern "C" void kernel_launch(void* const* d_in, const int* in_sizes, int n_in,
                              void* d_out, int out_size, void* d_ws, size_t ws_size,
                              hipStream_t stream)
{
  (void)n_in; (void)out_size; (void)ws_size;
  const int Bz = 32, S = 197, Dm = 768, Hn = 12, MLPd = 3072, L = 12;
  const int ROWS = Bz * S;        // 6304
  const int PROWS = Bz * 196;     // 6272

  // ---- input pointers (setup_inputs() dict order, leaves depth-first) ----
  const float* x       = (const float*)d_in[0];
  const float* w_patch = (const float*)d_in[1];
  const float* b_patch = (const float*)d_in[2];
  const float* cls     = (const float*)d_in[3];
  const float* pos     = (const float*)d_in[4];
  const float* ln1_g   = (const float*)d_in[5];
  const float* ln1_b   = (const float*)d_in[6];
  const float* wq      = (const float*)d_in[7];
  const float* bq      = (const float*)d_in[8];
  const float* wk      = (const float*)d_in[9];
  const float* bk      = (const float*)d_in[10];
  const float* wv_     = (const float*)d_in[11];
  const float* bv      = (const float*)d_in[12];
  const float* wo      = (const float*)d_in[13];
  const float* bo      = (const float*)d_in[14];
  const float* ln2_g   = (const float*)d_in[15];
  const float* ln2_b   = (const float*)d_in[16];
  const float* w1      = (const float*)d_in[17];
  const float* b1      = (const float*)d_in[18];
  const float* w2      = (const float*)d_in[19];
  const float* b2      = (const float*)d_in[20];
  const float* lnf_g   = (const float*)d_in[21];
  const float* lnf_b   = (const float*)d_in[22];
  const float* fc_W    = (const float*)d_in[23];
  const float* fc_b    = (const float*)d_in[24];
  const float* g1W = (const float*)d_in[25]; const float* g1as = (const float*)d_in[26];
  const float* g1ad = (const float*)d_in[27]; const float* g1b = (const float*)d_in[28];
  const float* g2W = (const float*)d_in[29]; const float* g2as = (const float*)d_in[30];
  const float* g2ad = (const float*)d_in[31]; const float* g2b = (const float*)d_in[32];
  const float* g3W = (const float*)d_in[33]; const float* g3as = (const float*)d_in[34];
  const float* g3ad = (const float*)d_in[35]; const float* g3b = (const float*)d_in[36];
  const int* esrc = (const int*)d_in[86];
  const int* edst = (const int*)d_in[87];
  const int E = in_sizes[86];     // 93

  UnetParams up;
  for (int i = 0; i < 5; ++i)  up.A[i]  = (const float*)d_in[37 + i];
  for (int i = 0; i < 10; ++i) { up.gW[i] = (const float*)d_in[42 + 2 * i]; up.gB[i] = (const float*)d_in[43 + 2 * i]; }
  for (int i = 0; i < 5; ++i)  { up.pW[i] = (const float*)d_in[62 + 2 * i]; up.pB[i] = (const float*)d_in[63 + 2 * i]; }
  for (int i = 0; i < 5; ++i)  { up.uW[i] = (const float*)d_in[72 + 2 * i]; up.uB[i] = (const float*)d_in[73 + 2 * i]; }
  up.fc1W = (const float*)d_in[82]; up.fc1b = (const float*)d_in[83];
  up.fc2W = (const float*)d_in[84]; up.fc2b = (const float*)d_in[85];

  // ---- workspace carve ----
  char* ws = (char*)d_ws;
  auto al = [](size_t v) { return (v + 255) & ~(size_t)255; };
  const size_t H768 = (size_t)ROWS * Dm;
  size_t off = 0;
  float* hbuf  = (float*)(ws + off); off = al(off + H768 * 4);
  float* qb    = (float*)(ws + off); off = al(off + H768 * 4);
  float* kb    = (float*)(ws + off); off = al(off + H768 * 4);
  float* vb    = (float*)(ws + off); off = al(off + H768 * 4);
  float* ob    = (float*)(ws + off); off = al(off + H768 * 4);
  float* g768  = (float*)(ws + off); off = al(off + H768 * 4);
  float* feats = (float*)(ws + off); off = al(off + (size_t)Bz * Dm * 4);
  float* gpre  = (float*)(ws + off); off = al(off + 928ull * 1024 * 4);
  float* gout  = (float*)(ws + off); off = al(off + 928ull * 1024 * 4);
  unsigned short* ybf    = (unsigned short*)(ws + off); off = al(off + H768 * 2);
  unsigned short* mlpbf  = (unsigned short*)(ws + off); off = al(off + (size_t)ROWS * MLPd * 2);
  unsigned short* wbf    = (unsigned short*)(ws + off); off = al(off + (size_t)Dm * MLPd * 2);
  unsigned short* nodebf = (unsigned short*)(ws + off); off = al(off + 928ull * 800 * 2);
  unsigned short* gbf    = (unsigned short*)(ws + off); off = al(off + 928ull * 1024 * 2);

  float* out = (float*)d_out;
  float* out_init = out;             // [32,29,2]
  float* out_2d   = out + 1856;      // [32,29,2]
  float* out_3d   = out + 3712;      // [32,29,3]

  // weight f32 [K,N] -> bf16 transposed [N,Kp]
  auto convT = [&](const float* s, unsigned short* d, int K, int N, int Kp) {
    dim3 g((Kp + 31) / 32, (N + 31) / 32);
    transpose_conv_kernel<<<g, 256, 0, stream>>>(s, d, K, N, Kp);
  };
  auto conv = [&](const float* s, unsigned short* d, size_t n) {   // n even
    size_t n2 = n >> 1;
    int g = (int)((n2 + 255) / 256); if (g > 2048) g = 2048;
    f32_to_bf16_kernel<<<g, 256, 0, stream>>>((const float2*)s, (unsigned int*)d, n2);
  };
  auto grid_of = [&](int M, int N) { return dim3((N + BN - 1) / BN, (M + BM - 1) / BM); };
  auto gemmF = [&](const unsigned short* A, const unsigned short* Bt, const float* bias,
                   float* C, int M, int N, int K) {
    gemm_bf16_kernel<0, 0, 1, 0><<<grid_of(M, N), 256, 0, stream>>>(A, Bt, bias, nullptr, C, M, N, K);
  };
  auto gemmResF = [&](const unsigned short* A, const unsigned short* Bt, const float* bias,
                      const float* R, float* C, int M, int N, int K) {
    gemm_bf16_kernel<0, 0, 1, 1><<<grid_of(M, N), 256, 0, stream>>>(A, Bt, bias, R, C, M, N, K);
  };
  auto gemmGeluBf = [&](const unsigned short* A, const unsigned short* Bt, const float* bias,
                        unsigned short* C, int M, int N, int K) {
    gemm_bf16_kernel<1, 1, 1, 0><<<grid_of(M, N), 256, 0, stream>>>(A, Bt, bias, nullptr, C, M, N, K);
  };
  auto gemmNB = [&](const unsigned short* A, const unsigned short* Bt, float* C,
                    int M, int N, int K) {
    gemm_bf16_kernel<0, 0, 0, 0><<<grid_of(M, N), 256, 0, stream>>>(A, Bt, nullptr, nullptr, C, M, N, K);
  };

  // ===== ViT patch embedding =====
  patchify_kernel<<<PROWS, 256, 0, stream>>>(x, ybf);
  convT(w_patch, wbf, Dm, Dm, Dm);
  gemmF(ybf, wbf, b_patch, g768, PROWS, Dm, Dm);
  assemble_h_kernel<<<ROWS, 256, 0, stream>>>(g768, cls, pos, hbuf);

  // ===== 12 transformer blocks =====
  for (int l = 0; l < L; ++l) {
    const size_t wOff = (size_t)l * Dm * Dm;
    ln_kernel<1><<<ROWS, 256, 0, stream>>>(hbuf, ln1_g + l * Dm, ln1_b + l * Dm, ybf, Dm);
    convT(wq + wOff, wbf, Dm, Dm, Dm);
    gemmF(ybf, wbf, bq + l * Dm, qb, ROWS, Dm, Dm);
    convT(wk + wOff, wbf, Dm, Dm, Dm);
    gemmF(ybf, wbf, bk + l * Dm, kb, ROWS, Dm, Dm);
    convT(wv_ + wOff, wbf, Dm, Dm, Dm);
    gemmF(ybf, wbf, bv + l * Dm, vb, ROWS, Dm, Dm);
    attn_kernel<<<dim3(Bz * Hn, (S + 7) / 8), 256, 0, stream>>>(qb, kb, vb, ob, S, Hn);
    conv(ob, ybf, H768);
    convT(wo + wOff, wbf, Dm, Dm, Dm);
    gemmResF(ybf, wbf, bo + l * Dm, hbuf, hbuf, ROWS, Dm, Dm);   // h += o @ Wo + bo
    ln_kernel<1><<<ROWS, 256, 0, stream>>>(hbuf, ln2_g + l * Dm, ln2_b + l * Dm, ybf, Dm);
    convT(w1 + (size_t)l * Dm * MLPd, wbf, Dm, MLPd, Dm);
    gemmGeluBf(ybf, wbf, b1 + (size_t)l * MLPd, mlpbf, ROWS, MLPd, Dm);
    convT(w2 + (size_t)l * MLPd * Dm, wbf, MLPd, Dm, MLPd);
    gemmResF(mlpbf, wbf, b2 + l * Dm, hbuf, hbuf, ROWS, Dm, MLPd); // h += mlp
  }

  // ===== final LN + token mean + head =====
  ln_kernel<0><<<ROWS, 256, 0, stream>>>(hbuf, lnf_g, lnf_b, g768, Dm);
  mean_kernel<<<Bz, 256, 0, stream>>>(g768, feats);
  head_kernel<<<8, 256, 0, stream>>>(feats, fc_W, fc_b, out_init);
  nodein_kernel<<<1024, 256, 0, stream>>>(out_init, feats, nodebf);

  // ===== GAT layers =====
  convT(g1W, wbf, 770, 1024, 800);
  gemmNB(nodebf, wbf, gpre, 928, 1024, 800);
  gat_agg_kernel<<<Bz, 256, 0, stream>>>(gpre, g1as, g1ad, g1b, esrc, edst, E, 4, 256, gout);
  conv(gout, gbf, 928ull * 1024);
  convT(g2W, wbf, 1024, 64, 1024);
  gemmNB(gbf, wbf, gpre, 928, 64, 1024);
  gat_agg_kernel<<<Bz, 256, 0, stream>>>(gpre, g2as, g2ad, g2b, esrc, edst, E, 4, 16, gout);
  conv(gout, gbf, 928ull * 64);
  convT(g3W, wbf, 64, 2, 64);
  gemmNB(gbf, wbf, gpre, 928, 2, 64);
  gat_agg_kernel<<<Bz, 256, 0, stream>>>(gpre, g3as, g3ad, g3b, esrc, edst, E, 1, 2, out_2d);

  // ===== Graph U-Net =====
  unet_kernel<<<Bz, 128, 0, stream>>>(out_2d, up, out_3d);
}
